// MambaAttention_75831942578811
// MI455X (gfx1250) — matrix operand order repaired and checked by reading, compiled, not run on hardware
//
#include <hip/hip_runtime.h>
#include <hip/hip_bf16.h>

// ---------------------------------------------------------------------------
// Problem constants (from reference)
// ---------------------------------------------------------------------------
#define B_    2
#define DM_   128
#define H__   64
#define W__   64
#define L_    4096          // H*W
#define DI_   256           // EXP*DM
#define DS_   16
#define DC_   4
#define DTR_  8
#define NR_   40            // DTR + 2*DS
#define BL_   (B_ * L_)     // 8192

typedef __attribute__((ext_vector_type(16))) _Float16 v16h;
typedef __attribute__((ext_vector_type(8)))  float    v8f;

// ---------------------------------------------------------------------------
// Direction permutation: scan index j -> original sequence position p.
// dir0: identity, dir1: flip, dir2: 64x64 transpose (involution).
// ---------------------------------------------------------------------------
__device__ __forceinline__ int perm_idx(int dir, int j) {
    if (dir == 0) return j;
    if (dir == 1) return (L_ - 1) - j;
    return ((j & 63) << 6) | (j >> 6);
}

// ---------------------------------------------------------------------------
// Generic WMMA f16 GEMM: C[M,N] = A[M,K] * B[K,N], f32 accumulate.
//   A: row-major (M,K), leading dim lda
//   B: BT==false -> row-major (K,N) ld ldb ; BT==true -> row-major (N,K) ld ldb
//   C: STORE_T==false -> row-major (M,N) ld ldc ; true -> (N,M) ld ldc
// One wave32 per 16x16 output tile; 2 waves per block. M,N mult of 16,
// K mult of 32.  EXEC stays all-ones inside each active wave (WMMA req).
// ---------------------------------------------------------------------------
template<bool BT, bool STORE_T>
__global__ __launch_bounds__(64)
void wmma_gemm_f16(const _Float16* __restrict__ A, int lda,
                   const _Float16* __restrict__ Bm, int ldb,
                   float* __restrict__ C, int ldc,
                   int M, int N, int K)
{
    const int lane  = threadIdx.x & 31;
    const int wave  = threadIdx.x >> 5;
    const int tiles_n = N >> 4;
    const int tiles   = (M >> 4) * tiles_n;
    const int tile = blockIdx.x * 2 + wave;
    if (tile >= tiles) return;                    // wave-uniform exit
    const int row0 = (tile / tiles_n) << 4;
    const int col0 = (tile % tiles_n) << 4;

    const int m   = lane & 15;
    const int n   = lane & 15;
    const int hid = lane >> 4;                    // lane-half: 0 or 1

    v8f acc = {};
    for (int k0 = 0; k0 < K; k0 += 32) {
        // ---- A fragment: 16x32, lanes 0-15 K=0..15, lanes 16-31 K=16..31 ----
        const _Float16* Arow = A + (size_t)(row0 + m) * lda + k0;
        const int kb = hid * 16;
        v16h a;
#pragma unroll
        for (int v = 0; v < 4; ++v) {
            a[2 * v]     = Arow[kb + 2 * v];
            a[2 * v + 1] = Arow[kb + 2 * v + 1];
            a[8 + 2 * v]     = Arow[8 + kb + 2 * v];
            a[8 + 2 * v + 1] = Arow[8 + kb + 2 * v + 1];
        }
        // ---- B fragment: 32x16, lanes 0-15 K=0..15, lanes 16-31 K=16..31 ----
        v16h b;
#pragma unroll
        for (int v = 0; v < 8; ++v) {
            const int k = k0 + hid * 16 + 2 * v;
            if (BT) {
                const _Float16* Bp = Bm + (size_t)(col0 + n) * ldb + k;
                b[2 * v]     = Bp[0];
                b[2 * v + 1] = Bp[1];
            } else {
                b[2 * v]     = Bm[(size_t)k       * ldb + col0 + n];
                b[2 * v + 1] = Bm[(size_t)(k + 1) * ldb + col0 + n];
            }
        }
        if (k0 + 32 < K) {                         // gfx1250 global_prefetch_b8
            __builtin_prefetch(Arow + 32, 0, 1);
        }
        acc = __builtin_amdgcn_wmma_f32_16x16x32_f16(
                  false, a, false, b, (short)0, acc, false, false);
    }

    // ---- store: C/D f32 16x16 layout (VGPR r -> M = r + 8*half) ----
    const int moff = hid * 8;
#pragma unroll
    for (int r = 0; r < 8; ++r) {
        if (STORE_T)
            C[(size_t)(col0 + n) * ldc + row0 + moff + r] = acc[r];
        else
            C[(size_t)(row0 + moff + r) * ldc + col0 + n] = acc[r];
    }
}

template<bool BT, bool STORE_T>
static inline void launch_gemm(const _Float16* A, int lda,
                               const _Float16* Bm, int ldb,
                               float* C, int ldc, int M, int N, int K,
                               hipStream_t s)
{
    const int tiles  = (M >> 4) * (N >> 4);
    const int blocks = (tiles + 1) >> 1;
    wmma_gemm_f16<BT, STORE_T><<<blocks, 64, 0, s>>>(A, lda, Bm, ldb, C, ldc, M, N, K);
}

// ---------------------------------------------------------------------------
// Elementwise helpers
// ---------------------------------------------------------------------------
__global__ void cvt_f16_kernel(const float* __restrict__ in,
                               _Float16* __restrict__ out, int nElem)
{
    int i = blockIdx.x * blockDim.x + threadIdx.x;
    if (i < nElem) out[i] = (_Float16)in[i];
}

__global__ void fill_zero_kernel(float* __restrict__ p, int nElem)
{
    int i = blockIdx.x * blockDim.x + threadIdx.x;
    if (i < nElem) p[i] = 0.f;
}

// ---------------------------------------------------------------------------
// LayerNorm over channels per (b,l); writes f16 row-major (BL, DM) for WMMA A.
// x layout: (B, DM, L).
// ---------------------------------------------------------------------------
__global__ __launch_bounds__(DM_)
void layernorm_kernel(const float* __restrict__ x,
                      const float* __restrict__ g,
                      const float* __restrict__ be,
                      _Float16* __restrict__ xn16)
{
    __shared__ float red[DM_];
    const int bl = blockIdx.x;
    const int b  = bl >> 12;
    const int l  = bl & (L_ - 1);
    const int c  = threadIdx.x;

    float v = x[((size_t)b * DM_ + c) * L_ + l];
    red[c] = v; __syncthreads();
    for (int o = DM_ / 2; o > 0; o >>= 1) {
        if (c < o) red[c] += red[c + o];
        __syncthreads();
    }
    const float mu = red[0] * (1.f / DM_);
    __syncthreads();
    const float dv = v - mu;
    red[c] = dv * dv; __syncthreads();
    for (int o = DM_ / 2; o > 0; o >>= 1) {
        if (c < o) red[c] += red[c + o];
        __syncthreads();
    }
    const float var = red[0] * (1.f / DM_);
    const float xn  = dv * rsqrtf(var + 1e-5f) * g[c] + be[c];
    xn16[(size_t)bl * DM_ + c] = (_Float16)xn;
}

// ---------------------------------------------------------------------------
// Causal depthwise conv (DC=4, left-pad 3) over the permuted sequence + SiLU.
// xz layout: (512, B*L)  (channel-major, x part = channels 0..255)
// xc output: (B, DI, L) in scan order.
// ---------------------------------------------------------------------------
__global__ void conv_silu_kernel(const float* __restrict__ xz,
                                 const float* __restrict__ cw,
                                 const float* __restrict__ cb,
                                 float* __restrict__ xc, int dir)
{
    int idx = blockIdx.x * blockDim.x + threadIdx.x;
    if (idx >= B_ * DI_ * L_) return;
    const int j = idx & (L_ - 1);
    const int d = (idx >> 12) & (DI_ - 1);
    const int b = idx >> 20;

    const float* w = cw + ((size_t)dir * DI_ + d) * DC_;
    float acc = cb[dir * DI_ + d];
#pragma unroll
    for (int t = 0; t < DC_; ++t) {
        const int jj = j - (DC_ - 1) + t;
        if (jj >= 0) {
            const int p = perm_idx(dir, jj);
            acc += w[t] * xz[(size_t)d * BL_ + b * L_ + p];
        }
    }
    const float s = acc / (1.f + __expf(-acc));     // SiLU
    xc[((size_t)b * DI_ + d) * L_ + j] = s;
}

// ---------------------------------------------------------------------------
// dbl[bl, r] = sum_d xc[b,d,j] * xproj_w[dir, r, d]      (r in 0..39)
// ---------------------------------------------------------------------------
__global__ void xproj_kernel(const float* __restrict__ xc,
                             const float* __restrict__ xpw,
                             float* __restrict__ dblb, int dir)
{
    int idx = blockIdx.x * blockDim.x + threadIdx.x;
    if (idx >= BL_ * NR_) return;
    const int r  = idx % NR_;
    const int bl = idx / NR_;
    const int b  = bl >> 12;
    const int j  = bl & (L_ - 1);

    const float* w  = xpw + ((size_t)dir * NR_ + r) * DI_;
    const float* xp = xc + (size_t)b * DI_ * L_ + j;
    float acc = 0.f;
    for (int d = 0; d < DI_; ++d) acc += xp[(size_t)d * L_] * w[d];
    dblb[(size_t)bl * NR_ + r] = acc;
}

// ---------------------------------------------------------------------------
// dt[bl, d] = softplus( dbl[bl, :8] . dtproj_w[dir, d, :] + dtproj_b[dir, d] )
// ---------------------------------------------------------------------------
__global__ void dtproj_kernel(const float* __restrict__ dblb,
                              const float* __restrict__ dpw,
                              const float* __restrict__ dpbv,
                              float* __restrict__ dtb, int dir)
{
    int idx = blockIdx.x * blockDim.x + threadIdx.x;
    if (idx >= BL_ * DI_) return;
    const int d  = idx & (DI_ - 1);
    const int bl = idx >> 8;

    const float* w = dpw + ((size_t)dir * DI_ + d) * DTR_;
    float acc = dpbv[dir * DI_ + d];
#pragma unroll
    for (int r = 0; r < DTR_; ++r) acc += dblb[(size_t)bl * NR_ + r] * w[r];
    const float sp = (acc > 20.f) ? acc : __logf(1.f + __expf(acc));
    dtb[(size_t)bl * DI_ + d] = sp;
}

// ---------------------------------------------------------------------------
// Selective scan: one thread per (b, d) channel; 16-state recurrence over
// L=4096 steps.  Fuses D*x skip, SiLU(z) gating and the inverse permutation
// scatter-add into ytot (layout (B*L, DI), original sequence order).
// ---------------------------------------------------------------------------
__global__ void scan_kernel(const float* __restrict__ xc,
                            const float* __restrict__ dtb,
                            const float* __restrict__ dblb,
                            const float* __restrict__ xz,
                            const float* __restrict__ Alog,
                            const float* __restrict__ Dv,
                            float* __restrict__ ytot, int dir)
{
    int idx = blockIdx.x * blockDim.x + threadIdx.x;
    if (idx >= B_ * DI_) return;
    const int b = idx >> 8;
    const int d = idx & (DI_ - 1);

    float negA[DS_];
#pragma unroll
    for (int n = 0; n < DS_; ++n)
        negA[n] = -__expf(Alog[((size_t)dir * DI_ + d) * DS_ + n]);
    const float Dd = Dv[dir * DI_ + d];

    float h[DS_];
#pragma unroll
    for (int n = 0; n < DS_; ++n) h[n] = 0.f;

    const float* xrow = xc + ((size_t)b * DI_ + d) * L_;
    const float* zrow = xz + (size_t)(DI_ + d) * BL_ + b * L_;

    for (int j = 0; j < L_; ++j) {
        const int m = b * L_ + j;
        const float dtv = dtb[(size_t)m * DI_ + d];
        const float xv  = xrow[j];
        const float* bc = dblb + (size_t)m * NR_ + DTR_;        // B (16)
        const float* cc = bc + DS_;                             // C (16)
        const float dx = dtv * xv;
        float y = 0.f;
#pragma unroll
        for (int n = 0; n < DS_; ++n) {
            h[n] = __expf(dtv * negA[n]) * h[n] + dx * bc[n];
            y += h[n] * cc[n];
        }
        y += Dd * xv;
        const int p = perm_idx(dir, j);
        const float z = zrow[p];
        const float g = z / (1.f + __expf(-z));                 // SiLU(z)
        ytot[(size_t)(b * L_ + p) * DI_ + d] += y * g;
    }
}

// ---------------------------------------------------------------------------
// Final batchnorm over (B,H,W) per channel, including the reference's
// reshape quirk: res(b,c,l) -> swapaxes(-1,-2) -> reshape(b,c,H,W), so
// out[b,c,t] comes from res[b, f&127, f>>7] with f = c*4096 + t.
// resb layout: (B, DM, L).
// ---------------------------------------------------------------------------
__global__ __launch_bounds__(256)
void batchnorm_kernel(const float* __restrict__ resb,
                      const float* __restrict__ g,
                      const float* __restrict__ be,
                      float* __restrict__ out)
{
    __shared__ float s1[256], s2[256];
    const int c   = blockIdx.x;
    const int tid = threadIdx.x;

    float sum = 0.f, sq = 0.f;
    for (int i = tid; i < B_ * L_; i += 256) {
        const int b = i >> 12;
        const int t = i & (L_ - 1);
        const int f = (c << 12) | t;
        const float v = resb[((size_t)b * DM_ + (f & 127)) * L_ + (f >> 7)];
        sum += v; sq += v * v;
    }
    s1[tid] = sum; s2[tid] = sq; __syncthreads();
    for (int o = 128; o > 0; o >>= 1) {
        if (tid < o) { s1[tid] += s1[tid + o]; s2[tid] += s2[tid + o]; }
        __syncthreads();
    }
    const float mu  = s1[0] * (1.f / (B_ * L_));
    const float var = s2[0] * (1.f / (B_ * L_)) - mu * mu;
    const float inv = rsqrtf(var + 1e-5f);
    const float gc  = g[c], bb = be[c];

    for (int i = tid; i < B_ * L_; i += 256) {
        const int b = i >> 12;
        const int t = i & (L_ - 1);
        const int f = (c << 12) | t;
        const float v = resb[((size_t)b * DM_ + (f & 127)) * L_ + (f >> 7)];
        out[((size_t)b * DM_ + c) * L_ + t] = (v - mu) * inv * gc + bb;
    }
}

// ---------------------------------------------------------------------------
// Host orchestration
// ---------------------------------------------------------------------------
extern "C" void kernel_launch(void* const* d_in, const int* in_sizes, int n_in,
                              void* d_out, int out_size, void* d_ws, size_t ws_size,
                              hipStream_t stream)
{
    (void)in_sizes; (void)n_in; (void)out_size; (void)ws_size;
    const float* x1   = (const float*)d_in[0];
    const float* x2   = (const float*)d_in[1];
    const float* ln_g = (const float*)d_in[2];
    const float* ln_b = (const float*)d_in[3];
    const float* ipw  = (const float*)d_in[4];   // (512,128)
    const float* cw   = (const float*)d_in[5];   // (3,256,1,4)
    const float* cb   = (const float*)d_in[6];   // (3,256)
    const float* xpw  = (const float*)d_in[7];   // (3,40,256)
    const float* dpw  = (const float*)d_in[8];   // (3,256,8)
    const float* dpb  = (const float*)d_in[9];   // (3,256)
    const float* Alog = (const float*)d_in[10];  // (3,256,16)
    const float* Dv   = (const float*)d_in[11];  // (3,256)
    const float* opw  = (const float*)d_in[12];  // (128,256)
    const float* bng  = (const float*)d_in[13];
    const float* bnb  = (const float*)d_in[14];
    float* out = (float*)d_out;

    char* wsp = (char*)d_ws;
    auto carve = [&](size_t bytes) -> void* {
        void* p = (void*)wsp;
        wsp += (bytes + 255) & ~(size_t)255;
        return p;
    };
    _Float16* xn16  = (_Float16*)carve((size_t)BL_ * DM_ * 2);
    _Float16* ipw16 = (_Float16*)carve((size_t)512 * DM_ * 2);
    _Float16* opw16 = (_Float16*)carve((size_t)DM_ * DI_ * 2);
    float*    xz    = (float*)  carve((size_t)512 * BL_ * 4);       // (512, B*L)
    float*    xc    = (float*)  carve((size_t)B_ * DI_ * L_ * 4);   // scan order
    float*    dblb  = (float*)  carve((size_t)BL_ * NR_ * 4);
    float*    dtb   = (float*)  carve((size_t)BL_ * DI_ * 4);
    float*    ytot  = (float*)  carve((size_t)BL_ * DI_ * 4);       // (B*L, DI)
    _Float16* y16   = (_Float16*)carve((size_t)BL_ * DI_ * 2);
    float*    Af    = (float*)  carve((size_t)DM_ * BL_ * 4);       // (128, B*L)
    float*    Bf    = (float*)  carve((size_t)DM_ * BL_ * 4);
    _Float16* Af16  = (_Float16*)carve((size_t)DM_ * BL_ * 2);
    _Float16* Bf16  = (_Float16*)carve((size_t)DM_ * BL_ * 2);
    float*    dpm   = (float*)  carve((size_t)B_ * DM_ * DM_ * 4);
    _Float16* dp16  = (_Float16*)carve((size_t)B_ * DM_ * DM_ * 2);
    float*    resb  = (float*)  carve((size_t)B_ * DM_ * L_ * 4);

    // Weight conversions (BT=true in the GEMMs means (N,K)-row-major weights
    // can be used directly -- no transpose needed).
    cvt_f16_kernel<<<(512 * DM_ + 255) / 256, 256, 0, stream>>>(ipw, ipw16, 512 * DM_);
    cvt_f16_kernel<<<(DM_ * DI_ + 255) / 256, 256, 0, stream>>>(opw, opw16, DM_ * DI_);

    auto run_feature = [&](const float* xin, float* feat) {
        // LN -> f16 A-matrix (B*L, 128)
        layernorm_kernel<<<BL_, DM_, 0, stream>>>(xin, ln_g, ln_b, xn16);
        // in_proj: (B*L,128) x (128,512) -> stored transposed as xz (512, B*L)
        launch_gemm<true, true>(xn16, DM_, ipw16, DM_, xz, BL_, BL_, 2 * DI_, DM_, stream);
        fill_zero_kernel<<<(BL_ * DI_ + 255) / 256, 256, 0, stream>>>(ytot, BL_ * DI_);
        for (int dir = 0; dir < 3; ++dir) {
            conv_silu_kernel<<<(B_ * DI_ * L_ + 255) / 256, 256, 0, stream>>>(xz, cw, cb, xc, dir);
            xproj_kernel<<<(BL_ * NR_ + 255) / 256, 256, 0, stream>>>(xc, xpw, dblb, dir);
            dtproj_kernel<<<(BL_ * DI_ + 255) / 256, 256, 0, stream>>>(dblb, dpw, dpb, dtb, dir);
            scan_kernel<<<2, 256, 0, stream>>>(xc, dtb, dblb, xz, Alog, Dv, ytot, dir);
        }
        cvt_f16_kernel<<<(BL_ * DI_ + 255) / 256, 256, 0, stream>>>(ytot, y16, BL_ * DI_);
        // out_proj: (B*L,256) x (256,128) -> stored transposed as feat (128, B*L)
        launch_gemm<true, true>(y16, DI_, opw16, DI_, feat, BL_, BL_, DM_, DI_, stream);
    };
    run_feature(x1, Af);
    run_feature(x2, Bf);

    cvt_f16_kernel<<<(DM_ * BL_ + 255) / 256, 256, 0, stream>>>(Af, Af16, DM_ * BL_);
    cvt_f16_kernel<<<(DM_ * BL_ + 255) / 256, 256, 0, stream>>>(Bf, Bf16, DM_ * BL_);

    // dp[b] = A_b (128x4096) * Bf_b^T (4096x128)   (BT: Bf rows are N)
    for (int b = 0; b < B_; ++b) {
        launch_gemm<true, false>(Af16 + (size_t)b * L_, BL_,
                                 Bf16 + (size_t)b * L_, BL_,
                                 dpm + (size_t)b * DM_ * DM_, DM_,
                                 DM_, DM_, L_, stream);
    }
    cvt_f16_kernel<<<(B_ * DM_ * DM_ + 255) / 256, 256, 0, stream>>>(dpm, dp16, B_ * DM_ * DM_);

    // res[b] = dp[b] (128x128) * Bf_b (128x4096)
    for (int b = 0; b < B_; ++b) {
        launch_gemm<false, false>(dp16 + (size_t)b * DM_ * DM_, DM_,
                                  Bf16 + (size_t)b * L_, BL_,
                                  resb + (size_t)b * DM_ * L_, L_,
                                  DM_, L_, DM_, stream);
    }

    batchnorm_kernel<<<DM_, 256, 0, stream>>>(resb, bng, bnb, out);
}
